// UpSampleLayer_17944373363244
// MI455X (gfx1250) — compile-verified, matrix operand order checked by reading, not run on hardware
//
#include <hip/hip_runtime.h>
#include <hip/hip_bf16.h>

// ---------------------------------------------------------------------------
// Problem constants:  N=8, Cin=outC=256, H=W=64, K=3,
// conv-transpose: 4x4 kernel, stride 2, pad 2 -> 128x128 output.
// ---------------------------------------------------------------------------
#define NB   8
#define CIN  256
#define OC   256
#define HH   64
#define WW   64
#define HW   4096           // 64*64
#define HWO  16384          // 128*128

typedef __attribute__((ext_vector_type(16))) __bf16 v16bf;
typedef __attribute__((ext_vector_type(8)))  float  v8f;

union Frag { unsigned int u[8]; v16bf v; };

__device__ __forceinline__ unsigned int bf16u(float f) {
    unsigned int b = __float_as_uint(f);
    return (b + 0x7FFFu + ((b >> 16) & 1u)) >> 16;   // round-to-nearest-even
}
__device__ __forceinline__ void setlo(Frag& f, uint4 v) {
    f.u[0] = v.x; f.u[1] = v.y; f.u[2] = v.z; f.u[3] = v.w;
}
__device__ __forceinline__ void sethi(Frag& f, uint4 v) {
    f.u[4] = v.x; f.u[5] = v.y; f.u[6] = v.z; f.u[7] = v.w;
}

// LDS row stride for the 16 x 256ch bf16 A tile: 128 u32 + 4 pad -> 528 B
#define AROW 132

// ---------------------------------------------------------------------------
// Kernel 0a: repack w_dcn (OC,CIN,3,3) f32 -> wt1[(k*OC+o)*CIN+c] bf16
// ---------------------------------------------------------------------------
__global__ __launch_bounds__(256)
void pack_wdcn_kernel(const float* __restrict__ wdcn, unsigned short* __restrict__ wt1)
{
    int i = blockIdx.x * 256 + threadIdx.x;          // 9*256*256
    if (i >= 9 * OC * CIN) return;
    int c = i & 255, o = (i >> 8) & 255, k = i >> 16;
    wt1[i] = (unsigned short)bf16u(wdcn[(o * CIN + c) * 9 + k]);
}

// ---------------------------------------------------------------------------
// Kernel 0b: repack w_ct (256,256,4,4) f32 -> wt2[((kr*4+ks)*OC+o)*CIN+c] bf16
//            where value = w_ct[c, o, kr, ks]
// ---------------------------------------------------------------------------
__global__ __launch_bounds__(256)
void pack_wct_kernel(const float* __restrict__ wct, unsigned short* __restrict__ wt2)
{
    int i = blockIdx.x * 256 + threadIdx.x;          // 16*256*256
    if (i >= 16 * OC * CIN) return;
    int c = i & 255, o = (i >> 8) & 255, krs = i >> 16;
    int kr = krs >> 2, ks = krs & 3;
    wt2[i] = (unsigned short)bf16u(wct[((c * OC + o) * 4 + kr) * 4 + ks]);
}

// ---------------------------------------------------------------------------
// Kernel 1: offset conv  om = conv3x3(x, w_off) + b_off
//   channels 0..17 -> off buffer, 18..26 -> sigmoid -> mask buffer
// ---------------------------------------------------------------------------
__global__ __launch_bounds__(256)
void offset_conv_kernel(const float* __restrict__ x,
                        const float* __restrict__ woff,
                        const float* __restrict__ boff,
                        float* __restrict__ off,
                        float* __restrict__ mask)
{
    int idx = blockIdx.x * blockDim.x + threadIdx.x;
    if (idx >= NB * HW) return;
    int n = idx >> 12, hw = idx & 4095, h = hw >> 6, w = hw & 63;

    float acc[27];
    #pragma unroll
    for (int ch = 0; ch < 27; ++ch) acc[ch] = boff[ch];

    for (int c = 0; c < CIN; ++c) {
        const float* xc = x + (((long)(n * CIN + c)) << 12);
        float xv[9];
        #pragma unroll
        for (int ki = 0; ki < 3; ++ki)
            #pragma unroll
            for (int kj = 0; kj < 3; ++kj) {
                int yy = h + ki - 1, xx = w + kj - 1;
                xv[ki * 3 + kj] =
                    (yy >= 0 && yy < HH && xx >= 0 && xx < WW) ? xc[(yy << 6) + xx] : 0.f;
            }
        const float* wc = woff + c * 9;
        #pragma unroll
        for (int ch = 0; ch < 27; ++ch) {
            const float* wp = wc + ch * (CIN * 9);
            float a = acc[ch];
            #pragma unroll
            for (int t = 0; t < 9; ++t) a += xv[t] * wp[t];
            acc[ch] = a;
        }
    }
    #pragma unroll
    for (int ch = 0; ch < 18; ++ch)
        off[(((long)(n * 18 + ch)) << 12) + hw] = acc[ch];
    #pragma unroll
    for (int k = 0; k < 9; ++k)
        mask[(((long)(n * 9 + k)) << 12) + hw] = 1.f / (1.f + __expf(-acc[18 + k]));
}

// ---------------------------------------------------------------------------
// Kernel 2: DCNv2 via WMMA.  One block (4 waves) per 16-pixel tile, covering
// ALL 256 output channels (wave handles 4 N-tiles).  The bilinear-sampled
// A tile (16px x 256ch bf16) is built once per tap in LDS and shared.
// grid = NB*HW/16
// ---------------------------------------------------------------------------
__global__ __launch_bounds__(128)
void dcn_wmma_kernel(const float* __restrict__ x,
                     const float* __restrict__ off,
                     const float* __restrict__ mask,
                     const unsigned short* __restrict__ wt1,  // (9, OC, CIN) bf16
                     const float* __restrict__ bdcn,
                     float* __restrict__ out1)                // (NB, OC, 64, 64)
{
    const int tid  = threadIdx.x;
    const int lane = tid & 31, wave = tid >> 5;
    const int row  = lane & 15;          // A row / B column index
    const int hi   = lane >> 4;          // K-half select

    const int hwb = (blockIdx.x * 16) & 4095;
    const int n   = (blockIdx.x * 16) >> 12;

    // sampling assignment: pixel sp (16), channel range [cs, cs+32)
    const int sp  = tid >> 3;
    const int cs  = (tid & 7) * 32;
    const int shw = hwb + sp;
    const int sh  = shw >> 6, sw = shw & 63;

    __shared__ unsigned int smem[16 * AROW];

    v8f zero = {};
    v8f acc[4];
    #pragma unroll
    for (int nt = 0; nt < 4; ++nt) acc[nt] = zero;

    #pragma unroll 1
    for (int k = 0; k < 9; ++k) {
        const int ki = k / 3, kj = k % 3;

        // ---- sampling phase: build bf16 A tile in LDS ----
        float oy = off[((n * 18 + 2 * k)     << 12) + shw];
        float ox = off[((n * 18 + 2 * k + 1) << 12) + shw];
        float mk = mask[((n * 9 + k) << 12) + shw];
        float ys = (float)(sh + ki - 1) + oy;
        float xs = (float)(sw + kj - 1) + ox;
        float y0f = floorf(ys), x0f = floorf(xs);
        float dy = ys - y0f, dx = xs - x0f;
        int y0 = (int)y0f, x0 = (int)x0f;

        float cw[4];
        int   cidx[4];
        #pragma unroll
        for (int c4 = 0; c4 < 4; ++c4) {
            int yy = y0 + (c4 >> 1), xx = x0 + (c4 & 1);
            bool v = (yy >= 0 && yy < HH && xx >= 0 && xx < WW);
            float fy = (c4 & 2) ? dy : (1.f - dy);
            float fx = (c4 & 1) ? dx : (1.f - dx);
            float wgt = fy * fx * mk;
            cw[c4] = v ? wgt : 0.f;                       // validity folded into weight
            int yc = yy < 0 ? 0 : (yy > 63 ? 63 : yy);
            int xc = xx < 0 ? 0 : (xx > 63 ? 63 : xx);
            cidx[c4] = ((n * CIN) << 12) + (yc << 6) + xc;
        }
        const unsigned int* wr0 = (const unsigned int*)0; (void)wr0;
        #pragma unroll
        for (int pp = 0; pp < 16; ++pp) {                 // 32 channels, paired
            int c = cs + 2 * pp;
            float v0 = 0.f, v1 = 0.f;
            #pragma unroll
            for (int c4 = 0; c4 < 4; ++c4) {              // unconditional clamped loads
                v0 += cw[c4] * x[cidx[c4] + (c << 12)];
                v1 += cw[c4] * x[cidx[c4] + ((c + 1) << 12)];
            }
            smem[sp * AROW + (cs >> 1) + pp] = bf16u(v0) | (bf16u(v1) << 16);
        }
        __syncthreads();

        // ---- compute phase: 8 K-chunks x 4 N-tiles ----
        const uint4* arow = (const uint4*)&smem[row * AROW];
        #pragma unroll 1
        for (int q = 0; q < 8; ++q) {
            Frag fa;
            setlo(fa, arow[q * 4 + hi]);                  // K = q*32 + hi*8 .. +7
            sethi(fa, arow[q * 4 + 2 + hi]);              // K = q*32 + 16 + hi*8 ..
            #pragma unroll
            for (int nt = 0; nt < 4; ++nt) {
                int ocol = wave * 64 + nt * 16 + row;
                const uint4* brow = (const uint4*)(wt1 + ((k * OC + ocol) << 8));
                Frag fb;
                setlo(fb, brow[q * 4 + hi]);
                sethi(fb, brow[q * 4 + 2 + hi]);
                acc[nt] = __builtin_amdgcn_wmma_f32_16x16x32_bf16(
                              false, fa.v, false, fb.v, (short)0, acc[nt], false, false);
            }
        }
        __syncthreads();
    }

    // epilogue: D element (M = r + 8*hi, N = row); add bias, store f32
    #pragma unroll
    for (int nt = 0; nt < 4; ++nt) {
        int ocol = wave * 64 + nt * 16 + row;
        float bias = bdcn[ocol];
        const long base = (((long)(n * OC + ocol)) << 12) + hwb;
        #pragma unroll
        for (int r = 0; r < 8; ++r)
            out1[base + r + 8 * hi] = acc[nt][r] + bias;
    }
}

// ---------------------------------------------------------------------------
// Kernel 3: per-channel batchnorm stats (one block per channel)
// ---------------------------------------------------------------------------
__global__ __launch_bounds__(256)
void bn_stats_kernel(const float* __restrict__ xin, float* __restrict__ mean,
                     float* __restrict__ inv, int per, int nimg)
{
    int ch = blockIdx.x, tid = threadIdx.x;
    float s = 0.f, s2 = 0.f;
    for (int nn = 0; nn < nimg; ++nn) {
        const float* p = xin + ((long)(nn * OC + ch)) * per;
        for (int i = tid; i < per; i += 256) {
            float v = p[i];
            s += v; s2 += v * v;
        }
    }
    __shared__ float sh[256], sh2[256];
    sh[tid] = s; sh2[tid] = s2;
    __syncthreads();
    for (int st = 128; st > 0; st >>= 1) {
        if (tid < st) { sh[tid] += sh[tid + st]; sh2[tid] += sh2[tid + st]; }
        __syncthreads();
    }
    if (tid == 0) {
        float cnt = (float)nimg * (float)per;
        float mu  = sh[0] / cnt;
        float var = sh2[0] / cnt - mu * mu;
        mean[ch] = mu;
        inv[ch]  = rsqrtf(var + 1e-5f);
    }
}

// ---------------------------------------------------------------------------
// Kernel 4: in-place BN apply + ReLU.  layout (n, 256, per), shift = log2(per)
// ---------------------------------------------------------------------------
__global__ __launch_bounds__(256)
void bn_apply_relu_kernel(float* __restrict__ xio, const float* __restrict__ mean,
                          const float* __restrict__ inv, const float* __restrict__ gamma,
                          const float* __restrict__ beta, int shift, long total)
{
    long i = (long)blockIdx.x * blockDim.x + threadIdx.x;
    if (i >= total) return;
    int ch = (int)((i >> shift) & 255);
    float v = (xio[i] - mean[ch]) * inv[ch] * gamma[ch] + beta[ch];
    xio[i] = v > 0.f ? v : 0.f;
}

// ---------------------------------------------------------------------------
// Kernel 5: conv-transpose (stride 2, 4x4, pad 2) via WMMA.
//   out[n,o,oy,ox] = sum_{c,r,s valid} xin[n,c,(oy-2+r)/2,(ox-2+s)/2] * wct[c,o,3-r,3-s]
//   Tiles have uniform output-column parity -> exactly 2x2 candidate taps.
//   One block per 16-pixel tile, all 256 out channels; A tile staged in LDS.
//   grid.x encodes (n, oy, px, t).
// ---------------------------------------------------------------------------
__global__ __launch_bounds__(128)
void convt_wmma_kernel(const float* __restrict__ xin,        // (NB,OC,64,64) post-BN
                       const unsigned short* __restrict__ wt2, // (16, OC, CIN) bf16
                       float* __restrict__ out)              // (NB, OC, 128, 128)
{
    const int tid  = threadIdx.x;
    const int lane = tid & 31, wave = tid >> 5;
    const int row  = lane & 15;
    const int hi   = lane >> 4;
    const int bx = blockIdx.x;
    const int t  = bx & 3;
    const int px = (bx >> 2) & 1;
    const int oy = (bx >> 3) & 127;
    const int n  = bx >> 10;

    // sampling assignment
    const int sp = tid >> 3;
    const int cs = (tid & 7) * 32;
    const int ox_s = px + 2 * (t * 16 + sp);

    __shared__ unsigned int smem[16 * AROW];

    v8f zero = {};
    v8f acc[4];
    #pragma unroll
    for (int nt = 0; nt < 4; ++nt) acc[nt] = zero;

    const int ry0 = oy & 1;
    #pragma unroll 1
    for (int rr = 0; rr < 2; ++rr) {
        int r  = ry0 + 2 * rr;
        int iy = (oy - 2 + r) >> 1;                      // even numerator
        if ((unsigned)iy >= (unsigned)HH) continue;      // uniform across block
        #pragma unroll 1
        for (int ss = 0; ss < 2; ++ss) {
            int s  = px + 2 * ss;

            // ---- sampling phase ----
            int ix = (ox_s - 2 + s) >> 1;
            bool av = ((unsigned)ix < (unsigned)WW);
            const float* src = xin + ((n * CIN) << 12) + (iy << 6) + (av ? ix : 0);
            #pragma unroll
            for (int pp = 0; pp < 16; ++pp) {
                int c = cs + 2 * pp;
                float v0 = av ? src[c << 12]       : 0.f;
                float v1 = av ? src[(c + 1) << 12] : 0.f;
                smem[sp * AROW + (cs >> 1) + pp] = bf16u(v0) | (bf16u(v1) << 16);
            }
            __syncthreads();

            // ---- compute phase ----
            const int krs = (3 - r) * 4 + (3 - s);
            const uint4* arow = (const uint4*)&smem[row * AROW];
            #pragma unroll 1
            for (int q = 0; q < 8; ++q) {
                Frag fa;
                setlo(fa, arow[q * 4 + hi]);
                sethi(fa, arow[q * 4 + 2 + hi]);
                #pragma unroll
                for (int nt = 0; nt < 4; ++nt) {
                    int ocol = wave * 64 + nt * 16 + row;
                    const uint4* brow = (const uint4*)(wt2 + ((krs * OC + ocol) << 8));
                    Frag fb;
                    setlo(fb, brow[q * 4 + hi]);
                    sethi(fb, brow[q * 4 + 2 + hi]);
                    acc[nt] = __builtin_amdgcn_wmma_f32_16x16x32_bf16(
                                  false, fa.v, false, fb.v, (short)0, acc[nt], false, false);
                }
            }
            __syncthreads();
        }
    }

    // epilogue: element (M = rI + 8*hi, N = row) -> ox_m = px + 2*(t*16+M)
    #pragma unroll
    for (int nt = 0; nt < 4; ++nt) {
        int ocol = wave * 64 + nt * 16 + row;
        const long base = (((long)(n * OC + ocol)) << 14) + ((long)oy << 7);
        #pragma unroll
        for (int rI = 0; rI < 8; ++rI) {
            int m   = rI + 8 * hi;
            int oxm = px + 2 * (t * 16 + m);
            out[base + oxm] = acc[nt][rI];
        }
    }
}

// ---------------------------------------------------------------------------
// Host launcher
// ---------------------------------------------------------------------------
extern "C" void kernel_launch(void* const* d_in, const int* in_sizes, int n_in,
                              void* d_out, int out_size, void* d_ws, size_t ws_size,
                              hipStream_t stream) {
    const float* x      = (const float*)d_in[0];
    const float* w_off  = (const float*)d_in[1];
    const float* b_off  = (const float*)d_in[2];
    const float* w_dcn  = (const float*)d_in[3];
    const float* b_dcn  = (const float*)d_in[4];
    const float* gamma1 = (const float*)d_in[5];
    const float* beta1  = (const float*)d_in[6];
    const float* w_ct   = (const float*)d_in[7];
    const float* gamma2 = (const float*)d_in[8];
    const float* beta2  = (const float*)d_in[9];
    float* out = (float*)d_out;

    // workspace layout: bf16 weight repacks first (16B aligned), then f32 bufs
    unsigned short* wt1 = (unsigned short*)d_ws;            // 9*256*256
    unsigned short* wt2 = wt1 + (size_t)9 * OC * CIN;       // 16*256*256
    float* off   = (float*)(wt2 + (size_t)16 * OC * CIN);   // NB*18*HW
    float* mask  = off   + (size_t)NB * 18 * HW;            // NB*9*HW
    float* out1  = mask  + (size_t)NB * 9 * HW;             // NB*OC*HW
    float* mean1 = out1  + (size_t)NB * OC * HW;
    float* inv1  = mean1 + OC;
    float* mean2 = inv1  + OC;
    float* inv2  = mean2 + OC;

    // 0. weight repacks (bf16, B-fragment-friendly [tap][o][c])
    pack_wdcn_kernel<<<(9 * OC * CIN + 255) / 256, 256, 0, stream>>>(w_dcn, wt1);
    pack_wct_kernel<<<(16 * OC * CIN + 255) / 256, 256, 0, stream>>>(w_ct, wt2);

    // 1. offset conv (+ sigmoid masks)
    offset_conv_kernel<<<(NB * HW + 255) / 256, 256, 0, stream>>>(
        x, w_off, b_off, off, mask);

    // 2. DCNv2 fused sample + WMMA GEMM (one block per 16-pixel tile)
    dcn_wmma_kernel<<<NB * HW / 16, 128, 0, stream>>>(
        x, off, mask, wt1, b_dcn, out1);

    // 3. BN1 stats + apply + ReLU (in place on out1)
    bn_stats_kernel<<<OC, 256, 0, stream>>>(out1, mean1, inv1, HW, NB);
    {
        long total = (long)NB * OC * HW;
        bn_apply_relu_kernel<<<(unsigned)((total + 255) / 256), 256, 0, stream>>>(
            out1, mean1, inv1, gamma1, beta1, 12, total);
    }
    // 4. conv-transpose via WMMA -> d_out
    convt_wmma_kernel<<<NB * 128 * 2 * 4, 128, 0, stream>>>(out1, wt2, out);

    // 5. BN2 stats + apply + ReLU (in place on d_out)
    bn_stats_kernel<<<OC, 256, 0, stream>>>(out, mean2, inv2, HWO, NB);
    {
        long total = (long)NB * OC * HWO;
        bn_apply_relu_kernel<<<(unsigned)((total + 255) / 256), 256, 0, stream>>>(
            out, mean2, inv2, gamma2, beta2, 14, total);
    }
}
